// EfficientWindowAttention_40724879900949
// MI455X (gfx1250) — matrix operand order repaired and checked by reading, not compiled, not run on hardware
//
#include <hip/hip_runtime.h>

typedef __bf16 bf16;
typedef __attribute__((ext_vector_type(16))) __bf16 bf16x16;
typedef __attribute__((ext_vector_type(8)))  __bf16 bf16x8;
typedef __attribute__((ext_vector_type(8)))  float  f32x8;

#define B_   2
#define S_   2048
#define D_   1024
#define H_   16
#define HD_  64
#define W_   256
#define SP_  2080   // S padded by 32 zero rows for window-edge tiles

// ---- fragment helpers (contiguous k order, k = 16*(lane/16) + s, wave32) ----
static __device__ __forceinline__ bf16x16 frag_cat(bf16x8 lo, bf16x8 hi) {
  union { bf16x16 v; bf16x8 h[2]; } u;
  u.h[0] = lo; u.h[1] = hi; return u.v;
}
// A (16xK, row-major source): lane<16 row=lane holds K 0..15; lane>=16 holds K 16..31
static __device__ __forceinline__ bf16x16 load_a_frag(const bf16* rowbase, int hi) {
  const bf16* p = rowbase + (hi ? 16 : 0);
  return frag_cat(*(const bf16x8*)p, *(const bf16x8*)(p + 8));
}
// B (Kx16, source contiguous in K per column): lane<16 col=lane holds K 0..15; lane>=16 K 16..31
static __device__ __forceinline__ bf16x16 load_b_frag(const bf16* colbase, int hi) {
  const bf16* p = colbase + (hi ? 16 : 0);
  return frag_cat(*(const bf16x8*)p, *(const bf16x8*)(p + 8));
}
#define WMMA_BF16(a,b,c) \
  __builtin_amdgcn_wmma_f32_16x16x32_bf16(false,(a),false,(b),(short)0,(c),false,false)

// ---- precision conversion kernels ----
__global__ void cvt_f32_bf16(const float* __restrict__ x, bf16* __restrict__ y, int n) {
  int i = blockIdx.x * blockDim.x + threadIdx.x;
  if (i < n) y[i] = (bf16)x[i];
}
// y[n*K + k] = (bf16) w[k*N + n]   (store weights transposed: contiguous-K per output column)
__global__ void cvt_transpose(const float* __restrict__ w, bf16* __restrict__ y, int K, int N) {
  int i = blockIdx.x * blockDim.x + threadIdx.x;
  if (i < K * N) { int k = i / N, n = i % N; y[(size_t)n * K + k] = (bf16)w[i]; }
}

// ---- QKV projection: [4096 x 1024] @ [1024 x 3072] + bias, scatter into Q/K/V^T bf16 ----
// Software-pipelined: fragments for k0+32 are loaded before the k0 WMMAs are issued,
// so waits become s_wait_loadcnt <= (in-flight prefetches) instead of 0.
__global__ void __launch_bounds__(128) qkv_gemm(const bf16* __restrict__ xbf,
    const bf16* __restrict__ wT, const float* __restrict__ bias,
    bf16* __restrict__ qbuf, bf16* __restrict__ kbuf, bf16* __restrict__ vT) {
  const int lane = threadIdx.x & 31;
  const int wave = threadIdx.x >> 5;
  const int m0   = blockIdx.x * 16;
  const int n0   = (blockIdx.y * 4 + wave) * 64;
  const int col  = lane & 15;
  const int hi   = lane >> 4;

  f32x8 acc0 = {}, acc1 = {}, acc2 = {}, acc3 = {};
  const bf16* arow  = xbf + (size_t)(m0 + col) * D_;
  const bf16* bcol0 = wT  + (size_t)(n0 + col) * D_;

  bf16x16 a_c  = load_a_frag(arow, hi);
  bf16x16 b_c0 = load_b_frag(bcol0,            hi);
  bf16x16 b_c1 = load_b_frag(bcol0 + 16 * D_, hi);
  bf16x16 b_c2 = load_b_frag(bcol0 + 32 * D_, hi);
  bf16x16 b_c3 = load_b_frag(bcol0 + 48 * D_, hi);

  for (int k0 = 0; k0 < D_; k0 += 32) {
    const int kn = (k0 + 32 < D_) ? (k0 + 32) : k0;      // last iter: harmless re-load
    bf16x16 a_n  = load_a_frag(arow + kn, hi);
    bf16x16 b_n0 = load_b_frag(bcol0 +            kn, hi);
    bf16x16 b_n1 = load_b_frag(bcol0 + 16 * D_ + kn, hi);
    bf16x16 b_n2 = load_b_frag(bcol0 + 32 * D_ + kn, hi);
    bf16x16 b_n3 = load_b_frag(bcol0 + 48 * D_ + kn, hi);

    acc0 = WMMA_BF16(a_c, b_c0, acc0);
    acc1 = WMMA_BF16(a_c, b_c1, acc1);
    acc2 = WMMA_BF16(a_c, b_c2, acc2);
    acc3 = WMMA_BF16(a_c, b_c3, acc3);

    a_c = a_n; b_c0 = b_n0; b_c1 = b_n1; b_c2 = b_n2; b_c3 = b_n3;
  }
#pragma unroll
  for (int s4 = 0; s4 < 4; ++s4) {
    f32x8 acc = (s4 == 0) ? acc0 : (s4 == 1) ? acc1 : (s4 == 2) ? acc2 : acc3;
    int c  = n0 + s4 * 16 + col;
    float bi = bias[c];
    int which = c >> 10;           // 0=q 1=k 2=v
    int cc = c & 1023;
    int h  = cc >> 6, hd = cc & 63;
#pragma unroll
    for (int r = 0; r < 8; ++r) {
      int m = m0 + hi * 8 + r;
      int b = m >> 11;             // / 2048
      int srow = m & 2047;
      bf16 v = (bf16)(acc[r] + bi);
      int bh = b * H_ + h;
      if (which == 0)      qbuf[(size_t)(bh * SP_ + srow) * HD_ + hd] = v;
      else if (which == 1) kbuf[(size_t)(bh * SP_ + srow) * HD_ + hd] = v;
      else                 vT[((size_t)bh * HD_ + hd) * SP_ + srow]   = v;
    }
  }
}

// ---- banded flash attention: one wave per (b,h,16-query tile) ----
// Pipelined: V fragments + next-tile K fragments are issued right after the score
// WMMAs; the softmax VALU stretch and the LDS P-transpose hide their latency.
__global__ void __launch_bounds__(128) attn_kernel(const bf16* __restrict__ qb,
    const bf16* __restrict__ kb, const bf16* __restrict__ vT, bf16* __restrict__ ao) {
  __shared__ __align__(16) bf16 pls[4][16 * 32];   // per-wave P-transpose scratch
  const int lane = threadIdx.x & 31;
  const int wave = threadIdx.x >> 5;
  const int tile = blockIdx.x * 4 + wave;          // 0..4095
  const int qt = tile & 127;
  const int bh = tile >> 7;                        // b*16 + h
  const int h  = bh & (H_ - 1);
  const int b  = bh >> 4;
  const int q0 = qt * 16;
  const int col = lane & 15;
  const int hi  = lane >> 4;

  const bf16* qrow = qb + (size_t)(bh * SP_ + q0 + col) * HD_;
  bf16x16 qf0 = load_a_frag(qrow,      hi);        // hd 0..31
  bf16x16 qf1 = load_a_frag(qrow + 32, hi);        // hd 32..63

  f32x8 o0 = {}, o1 = {}, o2 = {}, o3 = {};
  float mrow[8], lrow[8];
#pragma unroll
  for (int r = 0; r < 8; ++r) { mrow[r] = -__builtin_inff(); lrow[r] = 0.f; }

  int lo  = q0 - W_;        if (lo < 0) lo = 0;
  int hik = q0 + 15 + W_;   if (hik > S_ - 1) hik = S_ - 1;
  int kstart = lo & ~31;
  int nt = (hik - kstart + 32) >> 5;

  bf16* myp = pls[wave];
  const float sc = 0.03125f;                       // 1/sqrt(1024)
  const bf16* kbh = kb + (size_t)(bh * SP_ + col) * HD_;   // + key*HD_ per tile
  const bf16* vbh = vT + ((size_t)bh * HD_ + col) * SP_;   // + kbase per tile

  // prefetch K fragments for tile 0
  {
  }
  const bf16* kr0 = kbh + (size_t)kstart * HD_;
  bf16x16 kc0 = load_b_frag(kr0,                 hi);
  bf16x16 kc1 = load_b_frag(kr0 + 32,            hi);
  bf16x16 kc2 = load_b_frag(kr0 + 16 * HD_,      hi);
  bf16x16 kc3 = load_b_frag(kr0 + 16 * HD_ + 32, hi);

  for (int t = 0; t < nt; ++t) {
    int kbase = kstart + t * 32;                   // >= 0; < S_+32 (pad rows are zero)

    // scores with current (prefetched) K fragments
    f32x8 sa = {}, sb = {};
    sa = WMMA_BF16(qf0, kc0, sa);
    sa = WMMA_BF16(qf1, kc1, sa);
    sb = WMMA_BF16(qf0, kc2, sb);
    sb = WMMA_BF16(qf1, kc3, sb);

    // issue V loads for this tile (independent of the scores)
    const bf16* vrow = vbh + kbase;
    bf16x16 vf0 = load_b_frag(vrow,            hi);
    bf16x16 vf1 = load_b_frag(vrow + 16 * SP_, hi);
    bf16x16 vf2 = load_b_frag(vrow + 32 * SP_, hi);
    bf16x16 vf3 = load_b_frag(vrow + 48 * SP_, hi);

    // prefetch next tile's K fragments
    const int kbn = (t + 1 < nt) ? (kbase + 32) : kbase;
    const bf16* krn = kbh + (size_t)kbn * HD_;
    bf16x16 kn0 = load_b_frag(krn,                 hi);
    bf16x16 kn1 = load_b_frag(krn + 32,            hi);
    bf16x16 kn2 = load_b_frag(krn + 16 * HD_,      hi);
    bf16x16 kn3 = load_b_frag(krn + 16 * HD_ + 32, hi);

    // ---- online softmax (long VALU stretch hides the loads above) ----
    int ja = kbase + col, jb = ja + 16;
    float va[8], vb[8], rm[8];
#pragma unroll
    for (int r = 0; r < 8; ++r) {
      int i = q0 + hi * 8 + r;
      float xa = sa[r] * sc, xb = sb[r] * sc;
      bool oka = (ja < S_) && (ja >= i - W_) && (ja <= i + W_);
      bool okb = (jb < S_) && (jb >= i - W_) && (jb <= i + W_);
      va[r] = oka ? xa : -__builtin_inff();
      vb[r] = okb ? xb : -__builtin_inff();
      rm[r] = fmaxf(va[r], vb[r]);
    }
    // row-max across the 16 lanes holding this half's rows
#pragma unroll
    for (int msk = 1; msk < 16; msk <<= 1)
#pragma unroll
      for (int r = 0; r < 8; ++r) rm[r] = fmaxf(rm[r], __shfl_xor(rm[r], msk, 32));

    float rs[8];
#pragma unroll
    for (int r = 0; r < 8; ++r) {
      float mn  = fmaxf(mrow[r], rm[r]);
      float mu  = (mn == -__builtin_inff()) ? 0.f : mn;                 // fully-masked guard
      float scl = (mrow[r] == -__builtin_inff()) ? 0.f : __expf(mrow[r] - mn);
      mrow[r] = mn;
      float pa = __expf(va[r] - mu);
      float pb = __expf(vb[r] - mu);
      rs[r] = pa + pb;
      lrow[r] *= scl;
      o0[r] *= scl; o1[r] *= scl; o2[r] *= scl; o3[r] *= scl;
      int row = hi * 8 + r;
      myp[row * 32 + col]      = (bf16)pa;
      myp[row * 32 + col + 16] = (bf16)pb;
    }
#pragma unroll
    for (int msk = 1; msk < 16; msk <<= 1)
#pragma unroll
      for (int r = 0; r < 8; ++r) rs[r] += __shfl_xor(rs[r], msk, 32);
#pragma unroll
    for (int r = 0; r < 8; ++r) lrow[r] += rs[r];

    asm volatile("s_wait_dscnt 0" ::: "memory");     // P stores visible before re-read
    bf16x16 pf = load_a_frag(myp + col * 32, hi);    // P as 16x32 A-fragment

    o0 = WMMA_BF16(pf, vf0, o0);
    o1 = WMMA_BF16(pf, vf1, o1);
    o2 = WMMA_BF16(pf, vf2, o2);
    o3 = WMMA_BF16(pf, vf3, o3);

    kc0 = kn0; kc1 = kn1; kc2 = kn2; kc3 = kn3;      // rotate K double-buffer
  }

  float inv[8];
#pragma unroll
  for (int r = 0; r < 8; ++r) inv[r] = 1.0f / lrow[r];
  bf16* obase = ao + (size_t)(b * S_ + q0) * D_ + h * HD_ + col;
#pragma unroll
  for (int r = 0; r < 8; ++r) {
    size_t ro = (size_t)(hi * 8 + r) * D_;
    obase[ro +  0] = (bf16)(o0[r] * inv[r]);
    obase[ro + 16] = (bf16)(o1[r] * inv[r]);
    obase[ro + 32] = (bf16)(o2[r] * inv[r]);
    obase[ro + 48] = (bf16)(o3[r] * inv[r]);
  }
}

// ---- output projection: [4096 x 1024] @ [1024 x 1024] + bias -> fp32 d_out ----
__global__ void __launch_bounds__(128) out_gemm(const bf16* __restrict__ abf,
    const bf16* __restrict__ wT, const float* __restrict__ bias, float* __restrict__ out) {
  const int lane = threadIdx.x & 31;
  const int wave = threadIdx.x >> 5;
  const int m0   = blockIdx.x * 16;
  const int n0   = (blockIdx.y * 4 + wave) * 64;
  const int col  = lane & 15;
  const int hi   = lane >> 4;

  f32x8 acc0 = {}, acc1 = {}, acc2 = {}, acc3 = {};
  const bf16* arow  = abf + (size_t)(m0 + col) * D_;
  const bf16* bcol0 = wT  + (size_t)(n0 + col) * D_;

  bf16x16 a_c  = load_a_frag(arow, hi);
  bf16x16 b_c0 = load_b_frag(bcol0,            hi);
  bf16x16 b_c1 = load_b_frag(bcol0 + 16 * D_, hi);
  bf16x16 b_c2 = load_b_frag(bcol0 + 32 * D_, hi);
  bf16x16 b_c3 = load_b_frag(bcol0 + 48 * D_, hi);

  for (int k0 = 0; k0 < D_; k0 += 32) {
    const int kn = (k0 + 32 < D_) ? (k0 + 32) : k0;
    bf16x16 a_n  = load_a_frag(arow + kn, hi);
    bf16x16 b_n0 = load_b_frag(bcol0 +            kn, hi);
    bf16x16 b_n1 = load_b_frag(bcol0 + 16 * D_ + kn, hi);
    bf16x16 b_n2 = load_b_frag(bcol0 + 32 * D_ + kn, hi);
    bf16x16 b_n3 = load_b_frag(bcol0 + 48 * D_ + kn, hi);

    acc0 = WMMA_BF16(a_c, b_c0, acc0);
    acc1 = WMMA_BF16(a_c, b_c1, acc1);
    acc2 = WMMA_BF16(a_c, b_c2, acc2);
    acc3 = WMMA_BF16(a_c, b_c3, acc3);

    a_c = a_n; b_c0 = b_n0; b_c1 = b_n1; b_c2 = b_n2; b_c3 = b_n3;
  }
#pragma unroll
  for (int s4 = 0; s4 < 4; ++s4) {
    f32x8 acc = (s4 == 0) ? acc0 : (s4 == 1) ? acc1 : (s4 == 2) ? acc2 : acc3;
    int c = n0 + s4 * 16 + col;
    float bi = bias[c];
#pragma unroll
    for (int r = 0; r < 8; ++r) {
      int m = m0 + hi * 8 + r;
      out[(size_t)m * D_ + c] = acc[r] + bi;
    }
  }
}

// ---- workspace layout (bytes) ----
static const size_t OFF_XBF  = 0;                    // 4096*1024*2      = 8388608
static const size_t OFF_WQKV = 8388608;              // 3072*1024*2      = 6291456
static const size_t OFF_WOUT = 14680064;             // 1024*1024*2      = 2097152
static const size_t QKV_BYTES = (size_t)B_ * H_ * SP_ * HD_ * 2;   // 8519680
static const size_t OFF_Q    = 16777216;
static const size_t OFF_K    = OFF_Q + QKV_BYTES;    // 25296896
static const size_t OFF_V    = OFF_K + QKV_BYTES;    // 33816576
static const size_t OFF_AO   = OFF_V + QKV_BYTES;    // 42336256, end ~50.7 MB

extern "C" void kernel_launch(void* const* d_in, const int* in_sizes, int n_in,
                              void* d_out, int out_size, void* d_ws, size_t ws_size,
                              hipStream_t stream) {
  (void)in_sizes; (void)n_in; (void)out_size; (void)ws_size;
  const float* x     = (const float*)d_in[0];
  const float* w_qkv = (const float*)d_in[1];
  const float* b_qkv = (const float*)d_in[2];
  const float* w_out = (const float*)d_in[3];
  const float* b_out = (const float*)d_in[4];

  char* ws = (char*)d_ws;
  bf16* xbf   = (bf16*)(ws + OFF_XBF);
  bf16* wqkvT = (bf16*)(ws + OFF_WQKV);
  bf16* woutT = (bf16*)(ws + OFF_WOUT);
  bf16* qb    = (bf16*)(ws + OFF_Q);
  bf16* kb    = (bf16*)(ws + OFF_K);
  bf16* vT    = (bf16*)(ws + OFF_V);
  bf16* ao    = (bf16*)(ws + OFF_AO);

  // zero Q/K/V (covers the 32 zero-padded rows used by window-edge tiles)
  hipMemsetAsync(ws + OFF_Q, 0, 3 * QKV_BYTES, stream);

  const int nx = B_ * S_ * D_;          // 4194304
  cvt_f32_bf16<<<(nx + 255) / 256, 256, 0, stream>>>(x, xbf, nx);
  cvt_transpose<<<(D_ * 3 * D_ + 255) / 256, 256, 0, stream>>>(w_qkv, wqkvT, D_, 3 * D_);
  cvt_transpose<<<(D_ * D_ + 255) / 256, 256, 0, stream>>>(w_out, woutT, D_, D_);

  qkv_gemm<<<dim3(256, 12), 128, 0, stream>>>(xbf, wqkvT, b_qkv, qb, kb, vT);
  attn_kernel<<<1024, 128, 0, stream>>>(qb, kb, vT, ao);
  out_gemm<<<dim3(256, 4), 128, 0, stream>>>(ao, woutT, b_out, (float*)d_out);
}
